// CUTLASSGroupedLinearOptimized_9363028706406
// MI455X (gfx1250) — matrix operand order repaired and checked by reading, not compiled, not run on hardware
//
#include <hip/hip_runtime.h>

// ---- problem constants (match reference) ----
#define NEXP   8
#define INF_   2048
#define OUTF_  2048
#define NTOK_  16384

// ---- tiling ----
#define BM 64
#define BN 128
#define BK 32
#define LDW 40              // ushorts per LDS row (32 data + 8 pad) -> conflict-free b128
#define KITERS (INF_ / BK)  // 64

typedef __attribute__((ext_vector_type(16))) __bf16         v16bf;
typedef __attribute__((ext_vector_type(8)))  float          v8f;
typedef __attribute__((ext_vector_type(4)))  float          f32x4;
typedef __attribute__((ext_vector_type(8)))  unsigned short u16x8;
typedef __attribute__((ext_vector_type(4)))  unsigned short u16x4;
typedef int v4i_ __attribute__((vector_size(4 * sizeof(int))));  // matches builtin proto

// ---- feature probes ----
#if defined(__has_builtin)
#  if __has_builtin(__builtin_amdgcn_global_load_async_to_lds_b128)
#    define HAS_ASYNC 1
#  endif
#endif
#ifndef HAS_ASYNC
#  define HAS_ASYNC 0
#endif

__device__ __forceinline__ unsigned short bf16_rne(float x) {
  unsigned u = __float_as_uint(x);
  u += 0x7FFFu + ((u >> 16) & 1u);
  return (unsigned short)(u >> 16);
}

__device__ __forceinline__ void split4(f32x4 v, u16x4* h, u16x4* l) {
#pragma unroll
  for (int i = 0; i < 4; ++i) {
    float x = v[i];
    unsigned short hb = bf16_rne(x);
    float hf = __uint_as_float(((unsigned)hb) << 16);
    (*h)[i] = hb;
    (*l)[i] = bf16_rne(x - hf);
  }
}

// A fragment (16x32 bf16): lane group g: K = g*8+0..7 (elems 0-7), 16+g*8+0..7 (elems 8-15)
__device__ __forceinline__ v16bf frag_a(const unsigned short* p, int row, int g) {
  union { u16x8 h[2]; v16bf v; } u;
  u.h[0] = *(const u16x8*)(p + row * LDW + g * 8);
  u.h[1] = *(const u16x8*)(p + row * LDW + 16 + g * 8);
  return u.v;
}
// B fragment (32x16 bf16): lane group g: K = g*8+0..7 (elems 0-7), 16+g*8+0..7 (elems 8-15)
__device__ __forceinline__ v16bf frag_b(const unsigned short* p, int row, int g) {
  union { u16x8 h[2]; v16bf v; } u;
  u.h[0] = *(const u16x8*)(p + row * LDW + g * 8);
  u.h[1] = *(const u16x8*)(p + row * LDW + 16 + g * 8);
  return u.v;
}

__device__ __forceinline__ v8f wmma_bf16(v16bf a, v16bf b, v8f c) {
  return __builtin_amdgcn_wmma_f32_16x16x32_bf16(false, a, false, b, (short)0, c,
                                                 false, false);
}

// 16B global -> LDS copy; async-to-LDS on CDNA5, plain copy otherwise
__device__ __forceinline__ void cp16(const unsigned short* g, unsigned short* l) {
#if HAS_ASYNC
  __builtin_amdgcn_global_load_async_to_lds_b128(
      (__attribute__((address_space(1))) v4i_*)g,
      (__attribute__((address_space(3))) v4i_*)l, 0, 0);
#else
  *(u16x8*)l = *(const u16x8*)g;
#endif
}

template <int N>
__device__ __forceinline__ void wait_async() {
#if HAS_ASYNC
#  if defined(__has_builtin) && __has_builtin(__builtin_amdgcn_s_wait_asynccnt)
  __builtin_amdgcn_s_wait_asynccnt(N);
#  else
  asm volatile("s_wait_asynccnt %0" ::"n"(N) : "memory");
#  endif
#endif
}

// ------------------- expert bucketing prologue -------------------
__global__ void moe_init(int* cnt) {
  if (threadIdx.x < NEXP) cnt[threadIdx.x] = 0;
}

__global__ void moe_count(const int* __restrict__ ea, int* __restrict__ cnt) {
  int t = blockIdx.x * blockDim.x + threadIdx.x;
  if (t < NTOK_) atomicAdd(&cnt[ea[t] & (NEXP - 1)], 1);
}

__global__ void moe_scan(const int* __restrict__ cnt, int* seg, int* cur, int* tileOff) {
  if (threadIdx.x == 0) {
    int o = 0, to = 0;
    for (int e = 0; e < NEXP; ++e) {
      seg[e] = o; cur[e] = o; tileOff[e] = to;
      o  += cnt[e];
      to += (cnt[e] + BM - 1) / BM;
    }
    seg[NEXP] = o;
    tileOff[NEXP] = to;
  }
}

__global__ void moe_fill(const int* __restrict__ ea, int* __restrict__ cur,
                         int* __restrict__ perm) {
  int t = blockIdx.x * blockDim.x + threadIdx.x;
  if (t < NTOK_) {
    int pos = atomicAdd(&cur[ea[t] & (NEXP - 1)], 1);
    perm[pos] = t;
  }
}

// ------------------- one-time fp32 -> (hi,lo) bf16 plane split -------------------
__global__ __launch_bounds__(256)
void split_planes(const float* __restrict__ src, unsigned short* __restrict__ hp,
                  unsigned short* __restrict__ lp, int n4) {
  int i = blockIdx.x * blockDim.x + threadIdx.x;
  if (i < n4) {
    size_t i4 = (size_t)i * 4;
    f32x4 v = *(const f32x4*)(src + i4);
    u16x4 h, l;
    split4(v, &h, &l);
    *(u16x4*)(hp + i4) = h;
    *(u16x4*)(lp + i4) = l;
  }
}

// ------------------- fast grouped GEMM: pre-split planes + async LDS staging -----
__global__ __launch_bounds__(128)
void moe_gemm_fast(const unsigned short* __restrict__ Ah, const unsigned short* __restrict__ Al,
                   const unsigned short* __restrict__ Bh, const unsigned short* __restrict__ Bl,
                   const int* __restrict__ perm, const int* __restrict__ seg,
                   const int* __restrict__ tileOff, float* __restrict__ out) {
  __shared__ unsigned short sA[2][2][BM * LDW];  // [buf][plane]
  __shared__ unsigned short sB[2][2][BN * LDW];
  __shared__ int rowTok[BM];

  const int tileM = blockIdx.x;
  if (tileM >= tileOff[NEXP]) return;  // uniform per block

  int e = 0;
#pragma unroll
  for (int i = 0; i < NEXP - 1; ++i)
    if (tileM >= tileOff[i + 1]) e = i + 1;
  const int lt = tileM - tileOff[e];
  const int segStart = seg[e];
  const int segEnd   = seg[e + 1];
  const int rowBase  = segStart + lt * BM;

  const int t = threadIdx.x;
  if (t < BM) {
    int pi = rowBase + t;
    rowTok[t] = (pi < segEnd) ? perm[pi] : -1;
  }
  __syncthreads();

  const int nBase = blockIdx.y * BN;
  const int off = (t & 3) * 8;  // 16B chunk within the BK slice
  const int r0  = t >> 2;       // base row (32 rows per pass)

  // per-thread source pointers (wave-uniform async counts: clamp invalid tokens to 0)
  const unsigned short* aSrc[2][2];
#pragma unroll
  for (int p = 0; p < 2; ++p) {
    int tok = rowTok[r0 + 32 * p];
    size_t base = (size_t)(tok >= 0 ? tok : 0) * INF_ + off;
    aSrc[p][0] = Ah + base;
    aSrc[p][1] = Al + base;
  }
  const unsigned short* bSrc[4][2];
#pragma unroll
  for (int p = 0; p < 4; ++p) {
    size_t base = ((size_t)e * OUTF_ + nBase + r0 + 32 * p) * INF_ + off;
    bSrc[p][0] = Bh + base;
    bSrc[p][1] = Bl + base;
  }

  auto issue = [&](int buf, int kb) {  // 12 async b128 per thread (uniform per wave)
#pragma unroll
    for (int p = 0; p < 2; ++p)
#pragma unroll
      for (int pl = 0; pl < 2; ++pl)
        cp16(aSrc[p][pl] + kb, &sA[buf][pl][(r0 + 32 * p) * LDW + off]);
#pragma unroll
    for (int p = 0; p < 4; ++p)
#pragma unroll
      for (int pl = 0; pl < 2; ++pl)
        cp16(bSrc[p][pl] + kb, &sB[buf][pl][(r0 + 32 * p) * LDW + off]);
  };

  // wave mapping: 4 waves -> 2x2 grid of 32x64 wave tiles
  const int wave = t >> 5, lane = t & 31;
  const int wm = (wave >> 1) * 32;
  const int wn = (wave & 1) * 64;
  const int lrow = lane & 15;
  const int g = lane >> 4;

  v8f acc[2][4];
#pragma unroll
  for (int sm = 0; sm < 2; ++sm)
#pragma unroll
    for (int sn = 0; sn < 4; ++sn)
      acc[sm][sn] = (v8f){0.f, 0.f, 0.f, 0.f, 0.f, 0.f, 0.f, 0.f};

  issue(0, 0);
  for (int i = 0; i < KITERS; ++i) {
    if (i + 1 < KITERS) {
      issue((i + 1) & 1, (i + 1) * BK);
      wait_async<12>();  // in-order: current tile's 12 asyncs complete
    } else {
      wait_async<0>();
    }
    __syncthreads();

    const unsigned short* pAh = sA[i & 1][0];
    const unsigned short* pAl = sA[i & 1][1];
    const unsigned short* pBh = sB[i & 1][0];
    const unsigned short* pBl = sB[i & 1][1];

    v16bf afh[2], afl[2];
#pragma unroll
    for (int sm = 0; sm < 2; ++sm) {
      afh[sm] = frag_a(pAh, wm + sm * 16 + lrow, g);
      afl[sm] = frag_a(pAl, wm + sm * 16 + lrow, g);
    }
#pragma unroll
    for (int sn = 0; sn < 4; ++sn) {
      v16bf bfh = frag_b(pBh, wn + sn * 16 + lrow, g);
      v16bf bfl = frag_b(pBl, wn + sn * 16 + lrow, g);
#pragma unroll
      for (int sm = 0; sm < 2; ++sm) {
        acc[sm][sn] = wmma_bf16(afh[sm], bfh, acc[sm][sn]);
        acc[sm][sn] = wmma_bf16(afh[sm], bfl, acc[sm][sn]);
        acc[sm][sn] = wmma_bf16(afl[sm], bfh, acc[sm][sn]);
      }
    }
    __syncthreads();  // all reads of buf done before it is refilled next iter
  }

#pragma unroll
  for (int sm = 0; sm < 2; ++sm) {
#pragma unroll
    for (int r = 0; r < 8; ++r) {
      int m = wm + sm * 16 + r + 8 * g;
      int tok = rowTok[m];
      if (tok >= 0) {
        float* op = out + (size_t)tok * OUTF_ + nBase + wn + lrow;
#pragma unroll
        for (int sn = 0; sn < 4; ++sn) op[sn * 16] = acc[sm][sn][r];
      }
    }
  }
}

// ------------------- fallback GEMM (convert-in-loop, used if ws too small) -------
__global__ __launch_bounds__(128)
void moe_gemm_cvt(const float* __restrict__ x, const float* __restrict__ w,
                  const int* __restrict__ perm, const int* __restrict__ seg,
                  const int* __restrict__ tileOff, float* __restrict__ out) {
  __shared__ unsigned short sAh[BM * LDW];
  __shared__ unsigned short sAl[BM * LDW];
  __shared__ unsigned short sBh[BN * LDW];
  __shared__ unsigned short sBl[BN * LDW];
  __shared__ int rowTok[BM];

  const int tileM = blockIdx.x;
  if (tileM >= tileOff[NEXP]) return;

  int e = 0;
#pragma unroll
  for (int i = 0; i < NEXP - 1; ++i)
    if (tileM >= tileOff[i + 1]) e = i + 1;
  const int lt = tileM - tileOff[e];
  const int segStart = seg[e];
  const int segEnd   = seg[e + 1];
  const int rowBase  = segStart + lt * BM;

  const int t = threadIdx.x;
  if (t < BM) {
    int pi = rowBase + t;
    rowTok[t] = (pi < segEnd) ? perm[pi] : -1;
  }
  __syncthreads();

  const int acol = (t & 7) * 4;
  const float* aptr[4];
  int arow[4];
#pragma unroll
  for (int p = 0; p < 4; ++p) {
    int r = (t >> 3) + 16 * p;
    arow[p] = r;
    int tok = rowTok[r];
    aptr[p] = (tok >= 0) ? (x + (size_t)tok * INF_) : nullptr;
  }

  const float* wbase = w + (size_t)e * OUTF_ * INF_;
  const int nBase = blockIdx.y * BN;
  const int wave = t >> 5, lane = t & 31;
  const int wm = (wave >> 1) * 32;
  const int wn = (wave & 1) * 64;
  const int lrow = lane & 15;
  const int g = lane >> 4;

  v8f acc[2][4];
#pragma unroll
  for (int sm = 0; sm < 2; ++sm)
#pragma unroll
    for (int sn = 0; sn < 4; ++sn)
      acc[sm][sn] = (v8f){0.f, 0.f, 0.f, 0.f, 0.f, 0.f, 0.f, 0.f};

  for (int kb = 0; kb < INF_; kb += BK) {
    __syncthreads();
#pragma unroll
    for (int p = 0; p < 4; ++p) {
      f32x4 v = {0.f, 0.f, 0.f, 0.f};
      if (aptr[p]) v = *(const f32x4*)(aptr[p] + kb + acol);
      u16x4 h, l;
      split4(v, &h, &l);
      *(u16x4*)(sAh + arow[p] * LDW + acol) = h;
      *(u16x4*)(sAl + arow[p] * LDW + acol) = l;
    }
#pragma unroll
    for (int p = 0; p < 8; ++p) {
      int r = (t >> 3) + 16 * p;
      const float* bp = wbase + (size_t)(nBase + r) * INF_ + kb + acol;
      f32x4 v = *(const f32x4*)bp;
      u16x4 h, l;
      split4(v, &h, &l);
      *(u16x4*)(sBh + r * LDW + acol) = h;
      *(u16x4*)(sBl + r * LDW + acol) = l;
    }
    __syncthreads();

    v16bf afh[2], afl[2];
#pragma unroll
    for (int sm = 0; sm < 2; ++sm) {
      afh[sm] = frag_a(sAh, wm + sm * 16 + lrow, g);
      afl[sm] = frag_a(sAl, wm + sm * 16 + lrow, g);
    }
#pragma unroll
    for (int sn = 0; sn < 4; ++sn) {
      v16bf bfh = frag_b(sBh, wn + sn * 16 + lrow, g);
      v16bf bfl = frag_b(sBl, wn + sn * 16 + lrow, g);
#pragma unroll
      for (int sm = 0; sm < 2; ++sm) {
        acc[sm][sn] = wmma_bf16(afh[sm], bfh, acc[sm][sn]);
        acc[sm][sn] = wmma_bf16(afh[sm], bfl, acc[sm][sn]);
        acc[sm][sn] = wmma_bf16(afl[sm], bfh, acc[sm][sn]);
      }
    }
  }

#pragma unroll
  for (int sm = 0; sm < 2; ++sm) {
#pragma unroll
    for (int r = 0; r < 8; ++r) {
      int m = wm + sm * 16 + r + 8 * g;
      int tok = rowTok[m];
      if (tok >= 0) {
        float* op = out + (size_t)tok * OUTF_ + nBase + wn + lrow;
#pragma unroll
        for (int sn = 0; sn < 4; ++sn) op[sn * 16] = acc[sm][sn][r];
      }
    }
  }
}

// ------------------- host-side launcher -------------------
extern "C" void kernel_launch(void* const* d_in, const int* in_sizes, int n_in,
                              void* d_out, int out_size, void* d_ws, size_t ws_size,
                              hipStream_t stream) {
  const float* x  = (const float*)d_in[0];
  const float* w  = (const float*)d_in[1];
  const int*   ea = (const int*)d_in[2];
  float* out = (float*)d_out;

  int* wsi     = (int*)d_ws;
  int* cnt     = wsi;       // 8
  int* seg     = wsi + 8;   // 9
  int* cur     = wsi + 20;  // 8
  int* tileOff = wsi + 28;  // 9
  int* perm    = wsi + 64;  // NTOK_

  const size_t HDR    = 65792;                         // bytes (16B aligned)
  const size_t XELEMS = (size_t)NTOK_ * INF_;          // 33,554,432
  const size_t WELEMS = (size_t)NEXP * OUTF_ * INF_;   // 33,554,432
  const size_t NEED   = HDR + 2 * (XELEMS + WELEMS) * sizeof(unsigned short);

  moe_init<<<1, 32, 0, stream>>>(cnt);
  moe_count<<<NTOK_ / 256, 256, 0, stream>>>(ea, cnt);
  moe_scan<<<1, 32, 0, stream>>>(cnt, seg, cur, tileOff);
  moe_fill<<<NTOK_ / 256, 256, 0, stream>>>(ea, cur, perm);

  dim3 grid(NTOK_ / BM + NEXP, OUTF_ / BN);

  if (ws_size >= NEED) {
    unsigned short* AH = (unsigned short*)((char*)d_ws + HDR);
    unsigned short* AL = AH + XELEMS;
    unsigned short* BH = AL + XELEMS;
    unsigned short* BL = BH + WELEMS;
    int n4x = (int)(XELEMS / 4), n4w = (int)(WELEMS / 4);
    split_planes<<<(n4x + 255) / 256, 256, 0, stream>>>(x, AH, AL, n4x);
    split_planes<<<(n4w + 255) / 256, 256, 0, stream>>>(w, BH, BL, n4w);
    moe_gemm_fast<<<grid, 128, 0, stream>>>(AH, AL, BH, BL, perm, seg, tileOff, out);
  } else {
    moe_gemm_cvt<<<grid, 128, 0, stream>>>(x, w, perm, seg, tileOff, out);
  }
}
